// TransformerLayer_16183436771717
// MI455X (gfx1250) — compile-verified, matrix operand order checked by reading, not run on hardware
//
#include <hip/hip_runtime.h>
#include <hip/hip_bf16.h>

typedef __attribute__((ext_vector_type(16))) __bf16 v16bf;
typedef __attribute__((ext_vector_type(8)))  float  v8f;

#define WMMA_BF16(a,b,c) __builtin_amdgcn_wmma_f32_16x16x32_bf16(false,(a),false,(b),(short)0,(c),false,false)

static constexpr int Bb   = 2;
static constexpr int Ss   = 2048;
static constexpr int Dd   = 1024;
static constexpr int Hh   = 16;
static constexpr int Ee   = 8;
static constexpr int DH   = 64;
static constexpr int Ntok = Bb * Ss;          // 4096
static constexpr int CAP  = 614;              // int(1.2 * 4096 / 8)
static constexpr int CAPP = 624;              // padded to multiple of 16

// ---- WMMA fragment builders --------------------------------------------
// 16-bit A/B fragment layout (wave32): lane 0-15 -> M/N = lane, K-base 0;
// lane 16-31 -> M/N = lane-16, K-base 8. VGPRs 0-3 hold K=kb..kb+7 (2/VGPR),
// VGPRs 4-7 hold K=kb+16..kb+23. So per lane: 8 contiguous elems at +kb and
// 8 contiguous elems at +kb+16 relative to the fragment's K origin.

__device__ inline v16bf frag_bf16(const __bf16* p) {
  v16bf f;
#pragma unroll
  for (int i = 0; i < 8; i++) f[i] = p[i];
#pragma unroll
  for (int i = 0; i < 8; i++) f[8 + i] = p[16 + i];
  return f;
}

__device__ inline v16bf frag_f32(const float* p) {
  v16bf f;
#pragma unroll
  for (int i = 0; i < 8; i++) f[i] = (__bf16)p[i];
#pragma unroll
  for (int i = 0; i < 8; i++) f[8 + i] = (__bf16)p[16 + i];
  return f;
}

// ---- 1. QKV projection: X[4096,1024] x W^T -> bf16 Q,K (row-major), V^T ----
__global__ void qkv_kernel(const float* __restrict__ X,
                           const float* __restrict__ Wq, const float* __restrict__ bq,
                           const float* __restrict__ Wk, const float* __restrict__ bk,
                           const float* __restrict__ Wv, const float* __restrict__ bv,
                           __bf16* __restrict__ Qbf, __bf16* __restrict__ Kbf,
                           __bf16* __restrict__ Vt) {
  int bid  = blockIdx.x;
  int mat  = bid / (256 * 64);
  int rem  = bid % (256 * 64);
  int rt   = rem / 64;                 // row tile over B*S
  int ct   = rem % 64;                 // col tile over H*DH
  int lane = threadIdx.x & 31;
  int half = lane >> 4, mn = lane & 15, kb = half * 8;

  const float* W   = (mat == 0) ? Wq : ((mat == 1) ? Wk : Wv);
  const float* bia = (mat == 0) ? bq : ((mat == 1) ? bk : bv);
  const float* arow = X + (size_t)(rt * 16 + mn) * Dd;
  const float* brow = W + (size_t)(ct * 16 + mn) * Dd;

  v8f c = {};
  for (int k = 0; k < Dd; k += 32) {
    v16bf a = frag_f32(arow + k + kb);
    v16bf b = frag_f32(brow + k + kb);
    c = WMMA_BF16(a, b, c);
  }
  int   gc  = ct * 16 + mn;            // output col = h*64 + dh (C layout: n = lane&15)
  float bv_ = bia[gc];
  int   h   = gc >> 6, dh = gc & 63;
#pragma unroll
  for (int r = 0; r < 8; r++) {
    int   gr = rt * 16 + r + half * 8; // C layout: m = r + 8*(lane>=16)
    int   b_ = gr >> 11, s = gr & 2047;
    float v  = c[r] + bv_;
    if (mat == 2)
      Vt[((size_t)(b_ * Hh + h) * DH + dh) * Ss + s] = (__bf16)v;
    else {
      __bf16* dst = (mat == 0) ? Qbf : Kbf;
      dst[((size_t)(b_ * Hh + h) * Ss + s) * DH + dh] = (__bf16)v;
    }
  }
}

// ---- 2. Attention: per 16-query strip, full-row scores in LDS ------------
// Faithful to reference: masked (j>i) scores set to 0 (NOT -inf), scaled by
// 1/sqrt(D)=1/32, softmax over ALL 2048 entries, P*V over ALL keys.
__global__ void __launch_bounds__(128) attn_kernel(const __bf16* __restrict__ Qbf,
                                                   const __bf16* __restrict__ Kbf,
                                                   const __bf16* __restrict__ Vt,
                                                   float* __restrict__ hidden) {
  __shared__ float Psc[16 * Ss];       // 128 KB score strip
  __shared__ float rowinv[16];

  int bid  = blockIdx.x;
  int tile = bid & 127, h = (bid >> 7) & 15, b = bid >> 11;
  int wid  = threadIdx.x >> 5, lane = threadIdx.x & 31;
  int half = lane >> 4, mn = lane & 15, kb = half * 8;

  const __bf16* Qb = Qbf + (size_t)(b * Hh + h) * Ss * DH;
  const __bf16* Kb = Kbf + (size_t)(b * Hh + h) * Ss * DH;
  const __bf16* Vb = Vt  + (size_t)(b * Hh + h) * DH * Ss;
  int s0 = tile * 16;

  // Q A-fragments for this wave (dh 0..31 and 32..63)
  const __bf16* qrow = Qb + (size_t)(s0 + mn) * DH;
  v16bf qa0 = frag_bf16(qrow + kb);
  v16bf qa1 = frag_bf16(qrow + 32 + kb);

  // Phase 1: scores S[16, 2048] = Q K^T, mask+scale, to LDS
  for (int jt = wid; jt < 128; jt += 4) {
    int j0 = jt * 16;
    const __bf16* krow = Kb + (size_t)(j0 + mn) * DH;  // B lane = key col
    v16bf k0 = frag_bf16(krow + kb);
    v16bf k1 = frag_bf16(krow + 32 + kb);
    v8f c = {};
    c = WMMA_BF16(qa0, k0, c);
    c = WMMA_BF16(qa1, k1, c);
    int j = j0 + mn;
#pragma unroll
    for (int r = 0; r < 8; r++) {
      int m = r + half * 8;
      int i = s0 + m;
      Psc[m * Ss + j] = (j <= i) ? c[r] * 0.03125f : 0.0f;
    }
  }
  __syncthreads();

  // Phase 2: full-row softmax (zeros included), fold 1/sum into rowinv
  for (int rr = 0; rr < 4; rr++) {
    int m = wid * 4 + rr;
    float* row = Psc + m * Ss;
    float mx = -1e30f;
    for (int jj = lane; jj < Ss; jj += 32) mx = fmaxf(mx, row[jj]);
#pragma unroll
    for (int o = 16; o > 0; o >>= 1) mx = fmaxf(mx, __shfl_xor(mx, o, 32));
    float sum = 0.f;
    for (int jj = lane; jj < Ss; jj += 32) {
      float e = __expf(row[jj] - mx);
      row[jj] = e;
      sum += e;
    }
#pragma unroll
    for (int o = 16; o > 0; o >>= 1) sum += __shfl_xor(sum, o, 32);
    if (lane == 0) rowinv[m] = 1.f / sum;
  }
  __syncthreads();

  // Phase 3: out[16,64] = P x V; each wave one 16-wide dh tile
  int n0 = wid * 16;
  const __bf16* vrow = Vb + (size_t)(n0 + mn) * Ss;  // Vt row = dh col
  v8f c = {};
  for (int ks = 0; ks < Ss; ks += 32) {
    v16bf a  = frag_f32(Psc + mn * Ss + ks + kb);    // P from LDS -> bf16
    v16bf bb = frag_bf16(vrow + ks + kb);
    c = WMMA_BF16(a, bb, c);
  }
#pragma unroll
  for (int r = 0; r < 8; r++) {
    int m = r + half * 8;
    int s = s0 + m;
    hidden[((size_t)(b * Ss) + s) * Dd + h * DH + n0 + mn] = c[r] * rowinv[m];
  }
}

// ---- 3. Router: logits = x @ Wsw^T + bsw, softmax, top-1 ----------------
__global__ void route_kernel(const float* __restrict__ hidden, const float* __restrict__ Wsw,
                             const float* __restrict__ bsw, int* __restrict__ routes,
                             float* __restrict__ probs) {
  int n = blockIdx.x, lane = threadIdx.x & 31;
  float acc[Ee];
#pragma unroll
  for (int e = 0; e < Ee; e++) acc[e] = 0.f;
  const float* x = hidden + (size_t)n * Dd;
  for (int d = lane; d < Dd; d += 32) {
    float xv = x[d];
#pragma unroll
    for (int e = 0; e < Ee; e++) acc[e] += xv * Wsw[e * Dd + d];
  }
#pragma unroll
  for (int e = 0; e < Ee; e++) {
#pragma unroll
    for (int o = 16; o > 0; o >>= 1) acc[e] += __shfl_xor(acc[e], o, 32);
  }
  if (lane == 0) {
    float l[Ee], m = -1e30f;
#pragma unroll
    for (int e = 0; e < Ee; e++) { l[e] = acc[e] + bsw[e]; m = fmaxf(m, l[e]); }
    int arg = 0; float best = l[0];
#pragma unroll
    for (int e = 1; e < Ee; e++) if (l[e] > best) { best = l[e]; arg = e; }  // first-max
    float Z = 0.f;
#pragma unroll
    for (int e = 0; e < Ee; e++) Z += __expf(l[e] - m);
    routes[n] = arg;
    probs[n]  = __expf(best - m) / Z;
  }
}

// ---- 4. Ordered capacity scan (single wave, exact cumsum order) ---------
__global__ void scan_kernel(const int* __restrict__ routes, int* __restrict__ rowslot,
                            int* __restrict__ kept) {
  int lane = threadIdx.x & 31;
  const int CH = Ntok / 32;            // 128 tokens per lane, contiguous
  int cnt[Ee];
#pragma unroll
  for (int e = 0; e < Ee; e++) cnt[e] = 0;
  int base = lane * CH;
  for (int i = 0; i < CH; i++) cnt[routes[base + i]]++;
  int off[Ee];
#pragma unroll
  for (int e = 0; e < Ee; e++) {
    int inc = cnt[e];
    for (int d = 1; d < 32; d <<= 1) {
      int t = __shfl_up(inc, d, 32);
      if (lane >= d) inc += t;
    }
    off[e] = inc - cnt[e];             // exclusive prefix across lanes
  }
  for (int i = 0; i < CH; i++) {
    int n = base + i;
    int r = routes[n];
    int p = off[r]++;
    kept[n]    = (p < CAP) ? 1 : 0;
    rowslot[n] = r * CAPP + ((p < CAP) ? p : 0);
  }
}

// ---- 5. Dispatch kept tokens into padded expert buffer (bf16) -----------
__global__ void dispatch_kernel(const float* __restrict__ hidden, const int* __restrict__ rowslot,
                                const int* __restrict__ kept, __bf16* __restrict__ ein) {
  int n = blockIdx.x;
  if (!kept[n]) return;
  int rs = rowslot[n];
  for (int d = threadIdx.x; d < Dd; d += blockDim.x)
    ein[(size_t)rs * Dd + d] = (__bf16)hidden[(size_t)n * Dd + d];
}

// ---- 6. Expert GEMM: out[e] = in[e] x We[e]^T + be[e] -------------------
__global__ void expert_kernel(const __bf16* __restrict__ ein, const float* __restrict__ We,
                              const float* __restrict__ be, float* __restrict__ eout) {
  int bid  = blockIdx.x;
  int e    = bid / ((CAPP / 16) * 64);
  int rem  = bid % ((CAPP / 16) * 64);
  int rt   = rem / 64, ct = rem % 64;
  int lane = threadIdx.x & 31;
  int half = lane >> 4, mn = lane & 15, kb = half * 8;

  const __bf16* arow = ein + (size_t)(e * CAPP + rt * 16 + mn) * Dd;
  const float*  brow = We + ((size_t)e * Dd + ct * 16 + mn) * Dd;  // We[e][f][d]
  v8f c = {};
  for (int k = 0; k < Dd; k += 32) {
    v16bf a = frag_bf16(arow + k + kb);
    v16bf b = frag_f32(brow + k + kb);
    c = WMMA_BF16(a, b, c);
  }
  int   gc   = ct * 16 + mn;
  float bias = be[e * Dd + gc];
#pragma unroll
  for (int r = 0; r < 8; r++) {
    int row = e * CAPP + rt * 16 + r + half * 8;
    eout[(size_t)row * Dd + gc] = c[r] + bias;
  }
}

// ---- 7. Gather + route-prob scale + residual + LayerNorm ----------------
__global__ void final_kernel(const float* __restrict__ hidden, const float* __restrict__ eout,
                             const int* __restrict__ rowslot, const int* __restrict__ kept,
                             const float* __restrict__ probs, const float* __restrict__ embed,
                             const float* __restrict__ gamma, const float* __restrict__ beta,
                             float* __restrict__ out) {
  int n = blockIdx.x, lane = threadIdx.x & 31;
  const float* src = kept[n] ? (eout + (size_t)rowslot[n] * Dd)
                             : (hidden + (size_t)n * Dd);
  float p = probs[n];
  float s1 = 0.f, s2 = 0.f;
  for (int d = lane; d < Dd; d += 32) {
    float x2 = src[d] * p + embed[(size_t)n * Dd + d];
    s1 += x2;
    s2 += x2 * x2;
  }
#pragma unroll
  for (int o = 16; o > 0; o >>= 1) {
    s1 += __shfl_xor(s1, o, 32);
    s2 += __shfl_xor(s2, o, 32);
  }
  float mean = s1 / Dd;
  float var  = s2 / Dd - mean * mean;
  float inv  = rsqrtf(var + 1e-5f);
  for (int d = lane; d < Dd; d += 32) {
    float x2 = src[d] * p + embed[(size_t)n * Dd + d];
    out[(size_t)n * Dd + d] = (x2 - mean) * inv * gamma[d] + beta[d];
  }
}

// ---- launcher -----------------------------------------------------------
extern "C" void kernel_launch(void* const* d_in, const int* in_sizes, int n_in,
                              void* d_out, int out_size, void* d_ws, size_t ws_size,
                              hipStream_t stream) {
  (void)in_sizes; (void)n_in; (void)out_size; (void)ws_size;
  const float* embed = (const float*)d_in[0];
  const float* Wq    = (const float*)d_in[1];
  const float* bq    = (const float*)d_in[2];
  const float* Wk    = (const float*)d_in[3];
  const float* bk    = (const float*)d_in[4];
  const float* Wv    = (const float*)d_in[5];
  const float* bv    = (const float*)d_in[6];
  const float* We    = (const float*)d_in[7];
  const float* be    = (const float*)d_in[8];
  const float* Wsw   = (const float*)d_in[9];
  const float* bsw   = (const float*)d_in[10];
  const float* gamma = (const float*)d_in[11];
  const float* beta  = (const float*)d_in[12];

  char*  ws  = (char*)d_ws;
  size_t off = 0;
  auto alloc = [&](size_t bytes) -> void* {
    void* p = ws + off;
    off += (bytes + 255) & ~(size_t)255;
    return p;
  };
  __bf16* Qbf    = (__bf16*)alloc((size_t)Bb * Hh * Ss * DH * sizeof(__bf16));
  __bf16* Kbf    = (__bf16*)alloc((size_t)Bb * Hh * Ss * DH * sizeof(__bf16));
  __bf16* Vt     = (__bf16*)alloc((size_t)Bb * Hh * Ss * DH * sizeof(__bf16));
  float*  hidden = (float*)alloc((size_t)Ntok * Dd * sizeof(float));
  __bf16* ein    = (__bf16*)alloc((size_t)Ee * CAPP * Dd * sizeof(__bf16));
  float*  eout   = (float*)alloc((size_t)Ee * CAPP * Dd * sizeof(float));
  int*    routes = (int*)alloc(Ntok * sizeof(int));
  float*  probs  = (float*)alloc(Ntok * sizeof(float));
  int*    rowslot= (int*)alloc(Ntok * sizeof(int));
  int*    kept   = (int*)alloc(Ntok * sizeof(int));

  qkv_kernel<<<3 * 256 * 64, 32, 0, stream>>>(embed, Wq, bq, Wk, bk, Wv, bv, Qbf, Kbf, Vt);
  attn_kernel<<<Bb * Hh * (Ss / 16), 128, 0, stream>>>(Qbf, Kbf, Vt, hidden);
  route_kernel<<<Ntok, 32, 0, stream>>>(hidden, Wsw, bsw, routes, probs);
  scan_kernel<<<1, 32, 0, stream>>>(routes, rowslot, kept);
  dispatch_kernel<<<Ntok, 128, 0, stream>>>(hidden, rowslot, kept, ein);
  expert_kernel<<<Ee * (CAPP / 16) * 64, 32, 0, stream>>>(ein, We, be, eout);
  final_kernel<<<Ntok, 32, 0, stream>>>(hidden, eout, rowslot, kept, probs, embed,
                                        gamma, beta, (float*)d_out);
}
